// TimeMoeAttention_49469433315532
// MI455X (gfx1250) — compile-verified, matrix operand order checked by reading, not run on hardware
//
#include <hip/hip_runtime.h>

typedef __attribute__((ext_vector_type(16))) _Float16 v16h;
typedef __attribute__((ext_vector_type(8)))  _Float16 v8h;
typedef __attribute__((ext_vector_type(8)))  float    v8f;
typedef __attribute__((ext_vector_type(4)))  unsigned int u32x4;
typedef __attribute__((ext_vector_type(8)))  unsigned int u32x8;

#define B_   2
#define S_   2048
#define H_   2048
#define NH_  16
#define NKV_ 4
#define HD_  128

// ---------- WMMA helpers (gfx1250 wave32, 16x16x32 f16 -> f32) ----------

__device__ inline v8f wmma16(v16h a, v16h b, v8f c) {
  return __builtin_amdgcn_wmma_f32_16x16x32_f16(false, a, false, b, (short)0, c,
                                                false, false);
}

// 16x32 A/B fragment from row-major f16 (leading dim ld).
__device__ inline v16h load_ab_f16(const _Float16* base, int ld) {
  const int lane = threadIdx.x & 31;
  const _Float16* p = base + (size_t)(lane & 15) * ld + ((lane >> 4) << 3);
  v8h lo = *(const v8h*)p;
  v8h hi = *(const v8h*)(p + 16);
  v16h r;
#pragma unroll
  for (int i = 0; i < 8; ++i) { r[i] = lo[i]; r[i + 8] = hi[i]; }
  return r;
}

// Same fragment from f32 source (LDS scores), converted to f16 inline.
__device__ inline v16h load_ab_f32(const float* base, int ld) {
  const int lane = threadIdx.x & 31;
  const float* p = base + (size_t)(lane & 15) * ld + ((lane >> 4) << 3);
  v16h r;
#pragma unroll
  for (int i = 0; i < 8; ++i) {
    r[i]     = (_Float16)p[i];
    r[i + 8] = (_Float16)p[i + 16];
  }
  return r;
}

// ---- Tensor Data Mover: 2D f16 tile (rows x 128 elems) global -> LDS ----
// Pads 4 dwords after every 64 dwords (= one 128-elem f16 row), so the LDS
// tile has a 136-f16 row stride (bank-conflict-free b128 fragment loads).
__device__ inline void tdm_load_k64(unsigned lds_addr, unsigned long long gaddr) {
  u32x4 g0;
  g0.x = 1u;                                   // count=1, user descriptor
  g0.y = lds_addr;                             // LDS byte address
  g0.z = (unsigned)gaddr;                      // global addr [95:64]
  g0.w = ((unsigned)(gaddr >> 32) & 0x01FFFFFFu) | (2u << 30);  // [120:96], type=2
  u32x8 g1;
  g1.s0 = (1u << 16)        // data_size = 1 (2 bytes)
        | (1u << 20)        // pad_enable
        | (5u << 22)        // pad_interval: 64 dwords
        | (3u << 25);       // pad_amount: 4 dwords
  g1.s1 = 128u << 16;       // tensor_dim0 = 128 (bits 63:48)
  g1.s2 = 2048u << 16;      // tensor_dim1 = 2048 (low 16, bits 95:80)
  g1.s3 = 128u << 16;       // tile_dim0 = 128 (bits 127:112)
  g1.s4 = 64u;              // tile_dim1 = 64, tile_dim2 = 0
  g1.s5 = 128u;             // tensor_dim0_stride = 128
  g1.s6 = 0u;
  g1.s7 = 0u;
  asm volatile("tensor_load_to_lds %0, %1" :: "s"(g0), "s"(g1) : "memory");
}

// ---------------- Kernel 1: fused QKV GEMM + bias + RoPE ----------------

__global__ __launch_bounds__(256) void qkv_rope_kernel(
    const float* __restrict__ X, const int* __restrict__ pos_ids,
    const float* __restrict__ Wq, const float* __restrict__ bq,
    const float* __restrict__ Wk, const float* __restrict__ bk,
    const float* __restrict__ Wv, const float* __restrict__ bv,
    _Float16* __restrict__ Qr, _Float16* __restrict__ Kr,
    _Float16* __restrict__ VT) {
  const int tid  = threadIdx.x;
  const int wave = tid >> 5;
  const int lane = tid & 31;
  const int mtile = blockIdx.x;
  const int ntile = blockIdx.y;

  const float* W; const float* bias; int Wld, ncol0;
  if (ntile < 16)      { W = Wq; bias = bq; Wld = NH_ * HD_;  ncol0 = ntile * 128; }
  else if (ntile < 20) { W = Wk; bias = bk; Wld = NKV_ * HD_; ncol0 = (ntile - 16) * 128; }
  else                 { W = Wv; bias = bv; Wld = NKV_ * HD_; ncol0 = (ntile - 20) * 128; }

  __shared__ __align__(16) char smem[64 * 132 * 4];
  _Float16* As = (_Float16*)smem;        // [64][40]
  _Float16* Bs = As + 64 * 40;           // [128][40]
  float*    Cs = (float*)smem;           // [64][132] (after K loop)

  const int m0   = mtile * 64;
  const int wrow = (wave >> 1) * 16;
  const int wcol = (wave & 1) * 64;

  v8f acc[4] = {};

  for (int k0 = 0; k0 < H_; k0 += 32) {
    {
      int row = tid >> 2, c = (tid & 3) << 3;
      const float* src = X + (size_t)(m0 + row) * H_ + k0 + c;
      _Float16* dst = As + row * 40 + c;
#pragma unroll
      for (int i = 0; i < 8; ++i) dst[i] = (_Float16)src[i];
    }
    {
      int k = tid >> 3, n = (tid & 7) << 4;
      const float* src = W + (size_t)(k0 + k) * Wld + ncol0 + n;
#pragma unroll
      for (int i = 0; i < 16; ++i) Bs[(n + i) * 40 + k] = (_Float16)src[i];
    }
    __syncthreads();
    // preload all fragments, then issue WMMAs back-to-back
    v16h a = load_ab_f16(As + wrow * 40, 40);
    v16h bf[4];
#pragma unroll
    for (int nb = 0; nb < 4; ++nb)
      bf[nb] = load_ab_f16(Bs + (wcol + nb * 16) * 40, 40);
#pragma unroll
    for (int nb = 0; nb < 4; ++nb)
      acc[nb] = wmma16(a, bf[nb], acc[nb]);
    __syncthreads();
  }

  {
    int col = lane & 15, half = lane >> 4;
#pragma unroll
    for (int nb = 0; nb < 4; ++nb)
#pragma unroll
      for (int r = 0; r < 8; ++r)
        Cs[(wrow + r + 8 * half) * 132 + wcol + nb * 16 + col] = acc[nb][r];
  }
  __syncthreads();

  if (ntile < 20) {
    const bool isQ = ntile < 16;
#pragma unroll
    for (int i = 0; i < 16; ++i) {
      int idx = tid + i * 256;
      int row = idx >> 6, d = idx & 63;
      int gr = m0 + row;
      int b = gr >> 11, s = gr & (S_ - 1);
      float x1 = Cs[row * 132 + d]      + bias[ncol0 + d];
      float x2 = Cs[row * 132 + d + 64] + bias[ncol0 + d + 64];
      float pos = (float)pos_ids[gr];
      float ang = pos * __expf((float)d * -0.14391157f);  // 10000^(-d/64)
      float cv = __cosf(ang), sv = __sinf(ang);
      float o1 = x1 * cv - x2 * sv;
      float o2 = x2 * cv + x1 * sv;
      if (isQ) {
        size_t base = (((size_t)b * NH_ + ntile) * S_ + s) * HD_;
        Qr[base + d] = (_Float16)o1;  Qr[base + d + 64] = (_Float16)o2;
      } else {
        size_t base = (((size_t)b * NKV_ + (ntile - 16)) * S_ + s) * HD_;
        Kr[base + d] = (_Float16)o1;  Kr[base + d + 64] = (_Float16)o2;
      }
    }
  } else {
    int hv = ntile - 20;
#pragma unroll
    for (int i = 0; i < 32; ++i) {
      int idx = tid + i * 256;
      int d = idx >> 6, row = idx & 63;
      int gr = m0 + row;
      int b = gr >> 11, s = gr & (S_ - 1);
      float val = Cs[row * 132 + d] + bias[ncol0 + d];
      VT[(((size_t)b * NKV_ + hv) * HD_ + d) * S_ + s] = (_Float16)val;
    }
  }
}

// ---------------- Kernel 2: flash attention (GQA, causal) ----------------
// 64-row q-block per WG; K in 64-wide chunks, TDM-staged to LDS with
// double buffering; online softmax; PV via WMMA.

__global__ __launch_bounds__(256) void attn_kernel(
    const _Float16* __restrict__ Qr, const _Float16* __restrict__ Kr,
    const _Float16* __restrict__ VT, _Float16* __restrict__ AO) {
  const int tid  = threadIdx.x;
  const int wave = tid >> 5;
  const int lane = tid & 31;
  const int qblk = blockIdx.x;
  const int h    = blockIdx.y;
  const int b    = blockIdx.z;
  const int kvh  = h >> 2;
  const int q0   = qblk * 64;

  __shared__ float    Ss[64 * 68];        // 64x64 scores, pad 4 f32
  __shared__ _Float16 Ks[2][64 * 136];    // TDM-padded K tiles (double buffer)
  __shared__ float mrow[64], lrow[64], crow[64];

  const _Float16* Qbase = Qr + (((size_t)b * NH_ + h) * S_ + q0) * HD_;
  const _Float16* Kbase = Kr + ((size_t)b * NKV_ + kvh) * S_ * HD_;
  const _Float16* Vbase = VT + ((size_t)b * NKV_ + kvh) * HD_ * S_;

  if (tid < 64) { mrow[tid] = -3.0e38f; lrow[tid] = 0.0f; }

  const int wrow  = (wave >> 1) * 16;  // q rows handled by wave
  const int wcolS = (wave & 1) * 32;   // score cols (kpos) handled by wave
  const int wcolO = (wave & 1) * 64;   // output cols (d) handled by wave
  v8f acc[4] = {};

  v16h qf[4];
#pragma unroll
  for (int kk = 0; kk < 4; ++kk)
    qf[kk] = load_ab_f16(Qbase + wrow * HD_ + kk * 32, HD_);

  const int jmax = qblk;               // causal: kpos chunks of 64
  const float scale = 0.08838834764831845f;  // 1/sqrt(128)

  if (wave == 0)
    tdm_load_k64((unsigned)(size_t)&Ks[0][0],
                 (unsigned long long)(size_t)(Kbase));
  __syncthreads();

  for (int j = 0; j <= jmax; ++j) {
    const int kp0 = j * 64;
    const _Float16* Kt = &Ks[j & 1][0];

    if (wave == 0) {
      if (j < jmax) {
        tdm_load_k64((unsigned)(size_t)&Ks[(j + 1) & 1][0],
                     (unsigned long long)(size_t)(Kbase + (size_t)(kp0 + 64) * HD_));
        __builtin_amdgcn_s_wait_tensorcnt(1);   // tile j complete
      } else {
        __builtin_amdgcn_s_wait_tensorcnt(0);
      }
    }
    __syncthreads();

    // scores = Q @ K^T for this wave's 16x32 region (K from TDM LDS tile)
    v8f sc[2] = {};
#pragma unroll
    for (int kk = 0; kk < 4; ++kk) {
      v16h kb[2];
#pragma unroll
      for (int nb = 0; nb < 2; ++nb)
        kb[nb] = load_ab_f16(Kt + (wcolS + nb * 16) * 136 + kk * 32, 136);
#pragma unroll
      for (int nb = 0; nb < 2; ++nb)
        sc[nb] = wmma16(qf[kk], kb[nb], sc[nb]);
    }
    {
      int col = lane & 15, half = lane >> 4;
#pragma unroll
      for (int nb = 0; nb < 2; ++nb)
#pragma unroll
        for (int r = 0; r < 8; ++r)
          Ss[(wrow + r + 8 * half) * 68 + wcolS + nb * 16 + col] = sc[nb][r];
    }
    __syncthreads();

    { // online softmax: 4 threads per row, 16 cols each
      int row = tid >> 2, part = tid & 3;
      int qpos = q0 + row;
      float* srow = Ss + row * 68 + part * 16;
      float vals[16];
      float mx = -3.0e38f;
#pragma unroll
      for (int c = 0; c < 16; ++c) {
        int kpos = kp0 + part * 16 + c;
        float v = srow[c] * scale;
        v = (kpos <= qpos) ? v : -3.0e38f;
        vals[c] = v;
        mx = fmaxf(mx, v);
      }
      mx = fmaxf(mx, __shfl_xor(mx, 1));
      mx = fmaxf(mx, __shfl_xor(mx, 2));
      float m_old = mrow[row];
      float m_new = fmaxf(m_old, mx);
      float sum = 0.0f;
#pragma unroll
      for (int c = 0; c < 16; ++c) {
        float p = (vals[c] > -1.0e38f) ? __expf(vals[c] - m_new) : 0.0f;
        srow[c] = p;
        sum += p;
      }
      sum += __shfl_xor(sum, 1);
      sum += __shfl_xor(sum, 2);
      if (part == 0) {
        crow[row] = __expf(m_old - m_new);
        lrow[row] = lrow[row] * crow[row] + sum;
        mrow[row] = m_new;
      }
    }
    __syncthreads();

    { // rescale accumulators
      int half = lane >> 4;
#pragma unroll
      for (int nb = 0; nb < 4; ++nb)
#pragma unroll
        for (int r = 0; r < 8; ++r)
          acc[nb][r] *= crow[wrow + r + 8 * half];
    }
    // acc += P @ V^T (preload fragments, then back-to-back WMMAs)
#pragma unroll
    for (int kk = 0; kk < 2; ++kk) {
      v16h pf = load_ab_f32(Ss + wrow * 68 + kk * 32, 68);
      v16h vb[4];
#pragma unroll
      for (int nb = 0; nb < 4; ++nb)
        vb[nb] = load_ab_f16(Vbase + (size_t)(wcolO + nb * 16) * S_ + kp0 + kk * 32, S_);
#pragma unroll
      for (int nb = 0; nb < 4; ++nb)
        acc[nb] = wmma16(pf, vb[nb], acc[nb]);
    }
    __syncthreads();
  }

  {
    int col = lane & 15, half = lane >> 4;
#pragma unroll
    for (int nb = 0; nb < 4; ++nb)
#pragma unroll
      for (int r = 0; r < 8; ++r) {
        int row = wrow + r + 8 * half;
        int d   = wcolO + nb * 16 + col;
        float v = acc[nb][r] / lrow[row];
        AO[((size_t)b * S_ + q0 + row) * (NH_ * HD_) + h * HD_ + d] = (_Float16)v;
      }
  }
}

// ---------------- Kernel 3: output projection (attn @ Wo) ----------------
// 128x128 tiles, wave = 32x64 region -> 8 WMMAs per K-step.

__global__ __launch_bounds__(256) void oproj_kernel(
    const _Float16* __restrict__ A, const float* __restrict__ Wo,
    float* __restrict__ out) {
  const int tid  = threadIdx.x;
  const int wave = tid >> 5;
  const int lane = tid & 31;
  const int m0 = blockIdx.x * 128;
  const int n0 = blockIdx.y * 128;

  __shared__ __align__(16) _Float16 As[128 * 40];
  __shared__ __align__(16) _Float16 Bs[128 * 40];

  const int wrow = (wave >> 1) * 32;
  const int wcol = (wave & 1) * 64;
  v8f acc[2][4] = {};

  for (int k0 = 0; k0 < H_; k0 += 32) {
    {
      int row = tid >> 1, c = (tid & 1) << 4;
      const _Float16* src = A + (size_t)(m0 + row) * H_ + k0 + c;
      _Float16* dst = As + row * 40 + c;
#pragma unroll
      for (int i = 0; i < 16; ++i) dst[i] = src[i];
    }
    {
      int k = tid >> 3, n = (tid & 7) << 4;
      const float* src = Wo + (size_t)(k0 + k) * H_ + n0 + n;
#pragma unroll
      for (int i = 0; i < 16; ++i) Bs[(n + i) * 40 + k] = (_Float16)src[i];
    }
    __syncthreads();
    v16h a0 = load_ab_f16(As + wrow * 40, 40);
    v16h a1 = load_ab_f16(As + (wrow + 16) * 40, 40);
    v16h bf[4];
#pragma unroll
    for (int nb = 0; nb < 4; ++nb)
      bf[nb] = load_ab_f16(Bs + (wcol + nb * 16) * 40, 40);
#pragma unroll
    for (int nb = 0; nb < 4; ++nb) {
      acc[0][nb] = wmma16(a0, bf[nb], acc[0][nb]);
      acc[1][nb] = wmma16(a1, bf[nb], acc[1][nb]);
    }
    __syncthreads();
  }
  {
    int col = lane & 15, half = lane >> 4;
#pragma unroll
    for (int mi = 0; mi < 2; ++mi)
#pragma unroll
      for (int nb = 0; nb < 4; ++nb)
#pragma unroll
        for (int r = 0; r < 8; ++r)
          out[(size_t)(m0 + wrow + mi * 16 + r + 8 * half) * H_ +
              n0 + wcol + nb * 16 + col] = acc[mi][nb][r];
  }
}

// ------------------------------- launcher -------------------------------

extern "C" void kernel_launch(void* const* d_in, const int* in_sizes, int n_in,
                              void* d_out, int out_size, void* d_ws, size_t ws_size,
                              hipStream_t stream) {
  const float* X   = (const float*)d_in[0];
  // d_in[1] = attention_mask (causal; computed analytically in-kernel)
  const int*   pid = (const int*)d_in[2];
  const float* Wq  = (const float*)d_in[3];
  const float* bq  = (const float*)d_in[4];
  const float* Wk  = (const float*)d_in[5];
  const float* bk  = (const float*)d_in[6];
  const float* Wv  = (const float*)d_in[7];
  const float* bv  = (const float*)d_in[8];
  const float* Wo  = (const float*)d_in[9];
  float* out = (float*)d_out;

  char* ws = (char*)d_ws;
  const size_t qBytes  = (size_t)B_ * NH_  * S_ * HD_ * sizeof(_Float16);
  const size_t kvBytes = (size_t)B_ * NKV_ * S_ * HD_ * sizeof(_Float16);
  _Float16* Qr = (_Float16*)ws;
  _Float16* Kr = (_Float16*)(ws + qBytes);
  _Float16* VT = (_Float16*)(ws + qBytes + kvBytes);
  _Float16* AO = (_Float16*)(ws + qBytes + 2 * kvBytes);

  dim3 g1((B_ * S_) / 64, 24);
  qkv_rope_kernel<<<g1, 256, 0, stream>>>(X, pid, Wq, bq, Wk, bk, Wv, bv,
                                          Qr, Kr, VT);

  dim3 g2(S_ / 64, NH_, B_);
  attn_kernel<<<g2, 256, 0, stream>>>(Qr, Kr, VT, AO);

  dim3 g3((B_ * S_) / 128, H_ / 128);
  oproj_kernel<<<g3, 256, 0, stream>>>(AO, Wo, out);
}
